// GIN_30923764531499
// MI455X (gfx1250) — matrix operand order repaired and checked by reading, not compiled, not run on hardware
//
#include <hip/hip_runtime.h>
#include <hip/hip_bf16.h>

#define NNODES 100000
#define NEDGES 1600000
#define NHID   128
#define NCLASS 40

typedef __attribute__((ext_vector_type(16))) __bf16 v16bf;
typedef __attribute__((ext_vector_type(8)))  float  v8f;

// ---------------------------------------------------------------------------
// Pack f32 weight W[K][Nlog] (row-major) into bf16 B-operand tiles.
// Tile (kc,nc) covers K rows [kc*32, kc*32+32), cols [nc*16, nc*16+16).
// Contiguous-k order, not ISA table order (16-bit B 32x16): lane n<16 -> col n, K 0..15 packed;
// lane n+16 -> col n, K 16..31.  Stored lane-major: one contiguous 32-byte
// load per lane: out[tile*512 + lane*16 + e],
//   e=0..15 -> k = kc*32 + (lane>>4)*16 + e, n = nc*16 + (lane&15).
// Cols >= Nlog are zero-padded (lin2: 40 -> 48).
// ---------------------------------------------------------------------------
__global__ void pack_weights_kernel(const float* __restrict__ W,
                                    __bf16* __restrict__ out,
                                    int K, int Nlog, int ntiles) {
  int wid  = (blockIdx.x * blockDim.x + threadIdx.x) >> 5;
  int lane = threadIdx.x & 31;
  int total = (K >> 5) * ntiles;
  if (wid >= total) return;
  int kc = wid / ntiles;
  int nc = wid % ntiles;
  int hi = lane >> 4;
  int lr = lane & 15;
  int n  = nc * 16 + lr;
  __bf16* o = out + (size_t)wid * 512 + (size_t)lane * 16;
#pragma unroll
  for (int e = 0; e < 16; ++e) {
    int k = kc * 32 + hi * 16 + e;
    float v = (n < Nlog) ? W[(size_t)k * Nlog + n] : 0.0f;
    o[e] = (__bf16)v;
  }
}

// ---------------------------------------------------------------------------
// Edge scatter: one wave per edge; lane l handles features 4l..4l+3.
// agg[dst] += feat[src]   (f32 hardware atomics; agg stride is always 128)
// ---------------------------------------------------------------------------
__global__ void scatter_add_kernel(const float* __restrict__ feat, int fstride,
                                   const int* __restrict__ src,
                                   const int* __restrict__ dst,
                                   float* __restrict__ agg, int nedges) {
  int e    = (blockIdx.x * blockDim.x + threadIdx.x) >> 5;
  int lane = threadIdx.x & 31;
  if (e >= nedges) return;
  int s = src[e];
  int d = dst[e];
  const float4 v = *(const float4*)(feat + (size_t)s * fstride + lane * 4);
  float* a = agg + (size_t)d * NHID + lane * 4;
  unsafeAtomicAdd(a + 0, v.x);
  unsafeAtomicAdd(a + 1, v.y);
  unsafeAtomicAdd(a + 2, v.z);
  unsafeAtomicAdd(a + 3, v.w);
}

// ---------------------------------------------------------------------------
// Load one 16x32 A fragment (rows = 16 rows at arow base, k = kb..kb+31)
// in the contiguous-k A layout used by the packed B tiles, converting f32 -> bf16, optionally adding A2.
// lane (hi = lane>>4): elements  kb+hi*16 .. +16 (contiguous)
// ---------------------------------------------------------------------------
__device__ __forceinline__ v16bf load_a_frag(const float* __restrict__ arow,
                                             const float* __restrict__ arow2,
                                             int kb) {
  float4 f0 = *(const float4*)(arow + kb);
  float4 f1 = *(const float4*)(arow + kb + 4);
  float4 f2 = *(const float4*)(arow + kb + 8);
  float4 f3 = *(const float4*)(arow + kb + 12);
  if (arow2) {
    float4 g0 = *(const float4*)(arow2 + kb);
    float4 g1 = *(const float4*)(arow2 + kb + 4);
    float4 g2 = *(const float4*)(arow2 + kb + 8);
    float4 g3 = *(const float4*)(arow2 + kb + 12);
    f0.x += g0.x; f0.y += g0.y; f0.z += g0.z; f0.w += g0.w;
    f1.x += g1.x; f1.y += g1.y; f1.z += g1.z; f1.w += g1.w;
    f2.x += g2.x; f2.y += g2.y; f2.z += g2.z; f2.w += g2.w;
    f3.x += g3.x; f3.y += g3.y; f3.z += g3.z; f3.w += g3.w;
  }
  v16bf a;
  a[0]  = (__bf16)f0.x; a[1]  = (__bf16)f0.y; a[2]  = (__bf16)f0.z; a[3]  = (__bf16)f0.w;
  a[4]  = (__bf16)f1.x; a[5]  = (__bf16)f1.y; a[6]  = (__bf16)f1.z; a[7]  = (__bf16)f1.w;
  a[8]  = (__bf16)f2.x; a[9]  = (__bf16)f2.y; a[10] = (__bf16)f2.z; a[11] = (__bf16)f2.w;
  a[12] = (__bf16)f3.x; a[13] = (__bf16)f3.y; a[14] = (__bf16)f3.z; a[15] = (__bf16)f3.w;
  return a;
}

// ---------------------------------------------------------------------------
// WMMA GEMM:  out[r, c] = act( (A[r,:] (+A2[r,:])) @ Wp + bias[c] )
// Block = 256 threads = 8 waves; wave owns 32 rows x (TILES*16) cols
// (two 16-row A fragments share each LDS-staged B tile -> 2 WMMA per B load).
// Weights staged through a 32KB static LDS buffer in K-chunks of
// KCHUNK = 32/TILES k-steps; B reads are ds_load_b128 (separate pipe from
// the global A loads).  gridDim.y indexes column blocks of TILES tiles.
// ---------------------------------------------------------------------------
template <int TILES>
__global__ __launch_bounds__(256) void gemm_bf16_kernel(
    const float* __restrict__ A, const float* __restrict__ A2,
    const __bf16* __restrict__ Wp, const float* __restrict__ bias,
    float* __restrict__ out, int nrows, int K, int a_stride, int out_stride,
    int ncols, int ntiles, int relu) {
  constexpr int KCHUNK = 32 / TILES;  // k-steps staged per LDS fill (<=32KB)
  __shared__ __align__(16) char smem[32768];

  const int wave = threadIdx.x >> 5;
  const int lane = threadIdx.x & 31;
  const int hi   = lane >> 4;
  const int lr   = lane & 15;
  const int rowbase = (blockIdx.x * 8 + wave) * 32;
  const int nc0 = blockIdx.y * TILES;

  int r0 = rowbase + lr;       if (r0 > nrows - 1) r0 = nrows - 1;
  int r1 = rowbase + 16 + lr;  if (r1 > nrows - 1) r1 = nrows - 1;
  const float* arow0  = A + (size_t)r0 * a_stride;
  const float* arow1  = A + (size_t)r1 * a_stride;
  const float* arow0b = A2 ? (A2 + (size_t)r0 * NHID) : nullptr;
  const float* arow1b = A2 ? (A2 + (size_t)r1 * NHID) : nullptr;

  v8f acc0[TILES], acc1[TILES];
  const v8f zero = {0.f, 0.f, 0.f, 0.f, 0.f, 0.f, 0.f, 0.f};
#pragma unroll
  for (int t = 0; t < TILES; ++t) { acc0[t] = zero; acc1[t] = zero; }

  const int ksteps = K >> 5;
  for (int ks0 = 0; ks0 < ksteps; ks0 += KCHUNK) {
    const int kc = (ksteps - ks0 < KCHUNK) ? (ksteps - ks0) : KCHUNK;
    // --- cooperative stage of B tiles: kc * TILES tiles, 64 x 16B each ---
    for (int c = threadIdx.x; c < kc * TILES * 64; c += 256) {
      int tile   = c >> 6;
      int within = c & 63;
      int ks = ks0 + tile / TILES;
      int t  = tile % TILES;
      ((float4*)smem)[c] =
          ((const float4*)(Wp + ((size_t)(ks * ntiles + nc0 + t) << 9)))[within];
    }
    __syncthreads();

    for (int ki = 0; ki < kc; ++ki) {
      const int ks = ks0 + ki;
      const int kb = (ks << 5) + hi * 16;
      v16bf a0 = load_a_frag(arow0, arow0b, kb);
      v16bf a1 = load_a_frag(arow1, arow1b, kb);
#pragma unroll
      for (int t = 0; t < TILES; ++t) {
        const v16bf b = *(const v16bf*)(smem + ((ki * TILES + t) << 10) + (lane << 5));
        acc0[t] = __builtin_amdgcn_wmma_f32_16x16x32_bf16(
            false, a0, false, b, (short)0, acc0[t], false, false);
        acc1[t] = __builtin_amdgcn_wmma_f32_16x16x32_bf16(
            false, a1, false, b, (short)0, acc1[t], false, false);
      }
    }
    __syncthreads();
  }

  // Epilogue: C/D layout — VGPR v: lanes 0-15 -> row base+v, lanes 16-31 -> base+8+v
#pragma unroll
  for (int t = 0; t < TILES; ++t) {
    int col = (nc0 + t) * 16 + lr;
    if (col < ncols) {
      float bv = bias[col];
#pragma unroll
      for (int v = 0; v < 8; ++v) {
        int ra = rowbase + hi * 8 + v;
        int rb = rowbase + 16 + hi * 8 + v;
        if (ra < nrows) {
          float val = acc0[t][v] + bv;
          if (relu) val = fmaxf(val, 0.0f);
          out[(size_t)ra * out_stride + col] = val;
        }
        if (rb < nrows) {
          float val = acc1[t][v] + bv;
          if (relu) val = fmaxf(val, 0.0f);
          out[(size_t)rb * out_stride + col] = val;
        }
      }
    }
  }
}

// ---------------------------------------------------------------------------
// In-place log_softmax over 40 classes; one wave per row.
// Lane l covers col l; lanes 0..7 also cover col 32+l.
// ---------------------------------------------------------------------------
__global__ void logsoftmax_kernel(float* __restrict__ out, int nrows) {
  int row  = (blockIdx.x * blockDim.x + threadIdx.x) >> 5;
  int lane = threadIdx.x & 31;
  if (row >= nrows) return;
  float* r = out + (size_t)row * NCLASS;
  float v0 = r[lane];
  float v1 = (lane < 8) ? r[32 + lane] : -INFINITY;
  float m = fmaxf(v0, v1);
#pragma unroll
  for (int off = 16; off > 0; off >>= 1) m = fmaxf(m, __shfl_xor(m, off));
  float s = expf(v0 - m) + ((lane < 8) ? expf(v1 - m) : 0.0f);
#pragma unroll
  for (int off = 16; off > 0; off >>= 1) s += __shfl_xor(s, off);
  float ls = logf(s);
  r[lane] = v0 - m - ls;
  if (lane < 8) r[32 + lane] = v1 - m - ls;
}

// ---------------------------------------------------------------------------
extern "C" void kernel_launch(void* const* d_in, const int* in_sizes, int n_in,
                              void* d_out, int out_size, void* d_ws, size_t ws_size,
                              hipStream_t stream) {
  const float* x   = (const float*)d_in[0];
  const int*   ei  = (const int*)d_in[1];
  const int*   src = ei;
  const int*   dst = ei + NEDGES;
  const float* gc_w1[3] = {(const float*)d_in[2], (const float*)d_in[6],  (const float*)d_in[10]};
  const float* gc_b1[3] = {(const float*)d_in[3], (const float*)d_in[7],  (const float*)d_in[11]};
  const float* gc_w2[3] = {(const float*)d_in[4], (const float*)d_in[8],  (const float*)d_in[12]};
  const float* gc_b2[3] = {(const float*)d_in[5], (const float*)d_in[9],  (const float*)d_in[13]};
  const float* lin1_w = (const float*)d_in[14];
  const float* lin1_b = (const float*)d_in[15];
  const float* lin2_w = (const float*)d_in[16];
  const float* lin2_b = (const float*)d_in[17];

  // workspace layout
  char* ws = (char*)d_ws;
  const size_t sz_agg  = (size_t)NNODES * NHID * sizeof(float);      // 51.2 MB
  const size_t sz_tmp  = sz_agg;                                     // 51.2 MB
  const size_t sz_hcat = (size_t)NNODES * 3 * NHID * sizeof(float);  // 153.6 MB
  float* agg  = (float*)ws;
  float* tmp  = (float*)(ws + sz_agg);
  float* hcat = (float*)(ws + sz_agg + sz_tmp);
  float* hlin = (float*)(ws + sz_agg + sz_tmp + sz_hcat);
  __bf16* packed = (__bf16*)(ws + sz_agg + sz_tmp + 2 * sz_hcat);
  __bf16* wp_conv[3][2];
  for (int l = 0; l < 3; ++l) {
    wp_conv[l][0] = packed + (size_t)(2 * l + 0) * (NHID * NHID);
    wp_conv[l][1] = packed + (size_t)(2 * l + 1) * (NHID * NHID);
  }
  __bf16* wp_lin1 = packed + (size_t)6 * (NHID * NHID);
  __bf16* wp_lin2 = wp_lin1 + (size_t)(3 * NHID) * (3 * NHID);

  // --- pack all weights into bf16 B-tile layout ---
  {
    auto pack = [&](const float* W, __bf16* o, int K, int Nlog, int ntiles) {
      int waves = (K >> 5) * ntiles;
      int blocks = (waves + 7) / 8;
      hipLaunchKernelGGL(pack_weights_kernel, dim3(blocks), dim3(256), 0, stream,
                         W, o, K, Nlog, ntiles);
    };
    for (int l = 0; l < 3; ++l) {
      pack(gc_w1[l], wp_conv[l][0], NHID, NHID, NHID / 16);
      pack(gc_w2[l], wp_conv[l][1], NHID, NHID, NHID / 16);
    }
    pack(lin1_w, wp_lin1, 3 * NHID, 3 * NHID, (3 * NHID) / 16);
    pack(lin2_w, wp_lin2, 3 * NHID, NCLASS, 3);  // 40 cols padded to 48
  }

  const int rowblocks = (NNODES + 255) / 256;  // 8 waves x 32 rows per block
  const int scat_blocks = (NEDGES * 32 + 255) / 256;

  // --- 3 GIN conv layers ---
  for (int l = 0; l < 3; ++l) {
    const float* feat = (l == 0) ? x : (hcat + (size_t)(l - 1) * NHID);
    const int fstride = (l == 0) ? NHID : 3 * NHID;

    hipMemsetAsync(agg, 0, sz_agg, stream);
    hipLaunchKernelGGL(scatter_add_kernel, dim3(scat_blocks), dim3(256), 0, stream,
                       feat, fstride, src, dst, agg, NEDGES);

    // h = relu((feat + agg) @ w1 + b1)   -> tmp [N,128]
    hipLaunchKernelGGL((gemm_bf16_kernel<8>), dim3(rowblocks, 1), dim3(256), 0, stream,
                       feat, agg, wp_conv[l][0], gc_b1[l], tmp,
                       NNODES, NHID, fstride, NHID, NHID, NHID / 16, 1);
    // h_l = relu(tmp @ w2 + b2)          -> hcat[:, l*128 : (l+1)*128]
    hipLaunchKernelGGL((gemm_bf16_kernel<8>), dim3(rowblocks, 1), dim3(256), 0, stream,
                       tmp, (const float*)nullptr, wp_conv[l][1], gc_b2[l],
                       hcat + (size_t)l * NHID,
                       NNODES, NHID, NHID, 3 * NHID, NHID, NHID / 16, 1);
  }

  // --- lin1: relu(hcat @ lin1_w + b)  [N,384] -> hlin ---
  hipLaunchKernelGGL((gemm_bf16_kernel<8>), dim3(rowblocks, 3), dim3(256), 0, stream,
                     hcat, (const float*)nullptr, wp_lin1, lin1_b, hlin,
                     NNODES, 3 * NHID, 3 * NHID, 3 * NHID, 3 * NHID, (3 * NHID) / 16, 1);

  // --- lin2: hlin @ lin2_w + b -> d_out [N,40] (48 padded tiles, masked) ---
  hipLaunchKernelGGL((gemm_bf16_kernel<3>), dim3(rowblocks, 1), dim3(256), 0, stream,
                     hlin, (const float*)nullptr, wp_lin2, lin2_b, (float*)d_out,
                     NNODES, 3 * NHID, 3 * NHID, NCLASS, NCLASS, 3, 0);

  // --- log_softmax in place ---
  hipLaunchKernelGGL(logsoftmax_kernel, dim3((NNODES * 32 + 255) / 256), dim3(256),
                     0, stream, (float*)d_out, NNODES);
}
